// SCA_82789789597811
// MI455X (gfx1250) — matrix operand order repaired and checked by reading, not compiled, not run on hardware
//
#include <hip/hip_runtime.h>
#include <hip/hip_bf16.h>

// ---------------------------------------------------------------------------
// Shapes (fixed by the reference): b=8, E=384, NH=12, HD=32, H=W=64, L=4096,
// NC=16, LC=256, 3E=1152.
// ---------------------------------------------------------------------------
#define BATCH 8
#define EDIM  384
#define NHEAD 12
#define HDIM  32
#define LTOK  4096
#define NCHNK 16
#define LCHNK 256
#define E3    1152
#define SCALE_K 0.17677669529663687f // 1/sqrt(32)

typedef __bf16 bf16_t;
typedef __attribute__((ext_vector_type(16))) __bf16        v16bf;
typedef __attribute__((ext_vector_type(8)))  float         v8f;
typedef __attribute__((ext_vector_type(8)))  unsigned int  v8u;
typedef __attribute__((ext_vector_type(4)))  unsigned int  u32x4;
typedef __attribute__((ext_vector_type(8)))  int           i32x8;
typedef __attribute__((ext_vector_type(4)))  int           i32x4;

#if defined(__has_builtin)
#if __has_builtin(__builtin_amdgcn_tensor_load_to_lds) && \
    __has_builtin(__builtin_amdgcn_s_wait_tensorcnt)
#define HAVE_TDM 1
#endif
#endif

__device__ __forceinline__ float bf2f(unsigned short u) {
    unsigned int x = ((unsigned int)u) << 16;
    return __builtin_bit_cast(float, x);
}
__device__ __forceinline__ unsigned short f2bf(float f) {
    unsigned int u = __builtin_bit_cast(unsigned int, f);
    unsigned int r = u + 0x7fffu + ((u >> 16) & 1u); // round-to-nearest-even
    return (unsigned short)(r >> 16);
}

// A-fragment (16x32 bf16, row-major source [m][k], leading dim ld, 4B aligned):
// lane half 0 holds K 0..15, half 1 holds K 16..31; row = lane&15.
// Element pairs are K-contiguous -> 8 dword LDS loads.
__device__ __forceinline__ v16bf load_a_frag(const unsigned short* s, int ld) {
    int lane = threadIdx.x & 31;
    int row  = lane & 15;
    int kh   = (lane >> 4) * 16;
    const unsigned int* p = (const unsigned int*)(s + row * ld + kh);
    v8u raw;
#pragma unroll
    for (int j = 0; j < 8; ++j) raw[j] = p[j];
    return __builtin_bit_cast(v16bf, raw);
}

// B-fragment (32x16 bf16) from a [k][n] row-major source (strided u16 loads):
// lane half 0 holds K 0..15, half 1 holds K 16..31; col = lane&15.
__device__ __forceinline__ v16bf load_b_frag_kn(const unsigned short* s, int ld) {
    int lane = threadIdx.x & 31;
    int col  = lane & 15;
    int kh   = (lane >> 4) * 16;
    v16bf b;
#pragma unroll
    for (int e = 0; e < 16; ++e)
        b[e] = __builtin_bit_cast(bf16_t, s[(kh + e) * ld + col]);
    return b;
}

// B-fragment from a [n][k] source (K contiguous per lane -> 8 dword LDS loads).
__device__ __forceinline__ v16bf load_b_frag_nk(const unsigned short* s, int ld) {
    int lane = threadIdx.x & 31;
    int col  = lane & 15;
    int kh   = (lane >> 4) * 16;
    const unsigned int* p = (const unsigned int*)(s + col * ld + kh);
    v8u raw;
#pragma unroll
    for (int j = 0; j < 8; ++j) raw[j] = p[j];
    return __builtin_bit_cast(v16bf, raw);
}

__device__ __forceinline__ v8f wmma_bf16(v16bf a, v16bf b, v8f c) {
    return __builtin_amdgcn_wmma_f32_16x16x32_bf16(false, a, false, b,
                                                   (short)0, c, false, false);
}

#ifdef HAVE_TDM
// ---------------------------------------------------------------------------
// TDM: async 2D tile load (global -> LDS), bf16 elements (data_size = 2B).
// D# bit packing per CDNA5 ISA ch.10.8 (group0 128b, group1 256b).
// dims/strides in elements; addresses in bytes.  Issue from one wave only;
// completion gated by s_wait_tensorcnt.  This toolchain exposes the 6-arg
// builtin (g0, g1, g2, g3, extra_group, cpol); groups 2/3 and the extra group
// are zero (2D tensor).
// ---------------------------------------------------------------------------
__device__ __forceinline__ void tdm_load_2d_bf16(unsigned lds_addr, const void* gptr,
                                                 unsigned tile_x, unsigned tile_y,
                                                 unsigned tensor_x, unsigned tensor_y,
                                                 unsigned stride_x) {
    unsigned long long ga = (unsigned long long)(size_t)gptr;
    u32x4 g0;
    g0[0] = 1u;                                     // count=1, user descriptor
    g0[1] = lds_addr;                               // lds_addr (bytes)
    g0[2] = (unsigned)(ga & 0xffffffffu);           // global_addr[31:0]
    g0[3] = (unsigned)((ga >> 32) & 0x1ffffffu)     // global_addr[56:32]
            | (2u << 30);                           // type = 2 ("image")
    i32x8 g1;
    g1[0] = (int)(1u << 16);                        // wg_mask=0, data_size=1 (2B)
    g1[1] = (int)((tensor_x & 0xffffu) << 16);      // tensor_dim0[15:0] @ bits63:48
    g1[2] = (int)(((tensor_x >> 16) & 0xffffu)      // tensor_dim0[31:16]
            | ((tensor_y & 0xffffu) << 16));        // tensor_dim1[15:0]
    g1[3] = (int)(((tensor_y >> 16) & 0xffffu)      // tensor_dim1[31:16]
            | ((tile_x & 0xffffu) << 16));          // tile_dim0
    g1[4] = (int)(tile_y & 0xffffu);                // tile_dim1 (tile_dim2=0)
    g1[5] = (int)stride_x;                          // tensor_dim0_stride[31:0]
    g1[6] = 0;                                      // stride0[47:32], stride1[15:0]
    g1[7] = 0;                                      // stride1[47:16]
    i32x4 z4 = {0, 0, 0, 0};
    i32x8 z8 = {0, 0, 0, 0, 0, 0, 0, 0};
    __builtin_amdgcn_tensor_load_to_lds(g0, g1, z4, z4, z8, 0);
}
__device__ __forceinline__ unsigned lds_off32(const void* p) {
    return (unsigned)(size_t)p; // generic LDS address: low 32 bits = LDS offset
}
#endif

// ---------------------------------------------------------------------------
// f32 -> bf16 conversion
// ---------------------------------------------------------------------------
__global__ __launch_bounds__(256) void cvt_bf16_kernel(const float* __restrict__ src,
                                                       unsigned short* __restrict__ dst,
                                                       int n) {
    int i = blockIdx.x * 256 + threadIdx.x;
    if (i < n) dst[i] = f2bf(src[i]);
}

// ---------------------------------------------------------------------------
// Batched WMMA GEMM: C[b] = A(MxK) * B[b](KxN) + bias.  64x64 block tile,
// 8 waves (4M x 2N), each wave owns a 16x32 C strip.  K slabs of 32 are
// double-buffered in LDS via TDM async tile loads (fallback: coop copies).
// ---------------------------------------------------------------------------
template <bool OUTF32>
__global__ __launch_bounds__(256) void gemm_bf16_kernel(
    const unsigned short* __restrict__ A, const unsigned short* __restrict__ B,
    const float* __restrict__ bias, float* __restrict__ Cf,
    unsigned short* __restrict__ Cb, int M, int N, int K) {
    __shared__ unsigned short As[2][64 * 32];
    __shared__ unsigned short Bs[2][32 * 64];
    const int bidx = blockIdx.z;
    const unsigned short* Bb = B + (size_t)bidx * K * N;
    const int tid  = threadIdx.x;
    const int wave = tid >> 5;
    const int lane = tid & 31;
    const int wm = (wave >> 1) * 16; // 0,16,32,48
    const int wn = (wave & 1) * 32;  // 0,32
    const int m0 = blockIdx.x * 64;
    const int n0 = blockIdx.y * 64;
    const int nk = K >> 5;

    v8f c0 = {}, c1 = {};
#ifdef HAVE_TDM
    if (wave == 0) { // preload slab 0 into buffer 0
        tdm_load_2d_bf16(lds_off32(&As[0][0]), A + (size_t)m0 * K, 32, 64,
                         (unsigned)K, (unsigned)M, (unsigned)K);
        tdm_load_2d_bf16(lds_off32(&Bs[0][0]), Bb + n0, 64, 32,
                         (unsigned)N, (unsigned)K, (unsigned)N);
    }
    for (int ki = 0; ki < nk; ++ki) {
        const int buf = ki & 1;
        if (wave == 0) {
            if (ki + 1 < nk) { // issue next slab into other buffer, keep 2 in flight
                int k0n = (ki + 1) << 5;
                tdm_load_2d_bf16(lds_off32(&As[buf ^ 1][0]),
                                 A + (size_t)m0 * K + k0n, 32, 64,
                                 (unsigned)K, (unsigned)M, (unsigned)K);
                tdm_load_2d_bf16(lds_off32(&Bs[buf ^ 1][0]),
                                 Bb + (size_t)k0n * N + n0, 64, 32,
                                 (unsigned)N, (unsigned)K, (unsigned)N);
                __builtin_amdgcn_s_wait_tensorcnt(2); // current pair landed
            } else {
                __builtin_amdgcn_s_wait_tensorcnt(0);
            }
        }
        __syncthreads();
        v16bf a  = load_a_frag(&As[buf][wm * 32], 32);
        v16bf b0 = load_b_frag_kn(&Bs[buf][wn], 64);
        v16bf b1 = load_b_frag_kn(&Bs[buf][wn + 16], 64);
        c0 = wmma_bf16(a, b0, c0);
        c1 = wmma_bf16(a, b1, c1);
        __syncthreads(); // buffer free for the TDM issue of the next iteration
    }
#else
    for (int ki = 0; ki < nk; ++ki) {
        const int k0 = ki << 5;
#pragma unroll
        for (int i = tid; i < 64 * 32; i += 256) {
            int r = i >> 5, c = i & 31;
            As[0][i] = A[(size_t)(m0 + r) * K + k0 + c];
        }
#pragma unroll
        for (int i = tid; i < 32 * 64; i += 256) {
            int r = i >> 6, c = i & 63;
            Bs[0][i] = Bb[(size_t)(k0 + r) * N + n0 + c];
        }
        if (k0 + 32 < K) {
            __builtin_prefetch(&A[(size_t)(m0 + (tid & 63)) * K + k0 + 32], 0, 0);
            __builtin_prefetch(&Bb[(size_t)(k0 + 32 + (tid & 31)) * N + n0], 0, 0);
        }
        __syncthreads();
        v16bf a  = load_a_frag(&As[0][wm * 32], 32);
        v16bf b0 = load_b_frag_kn(&Bs[0][wn], 64);
        v16bf b1 = load_b_frag_kn(&Bs[0][wn + 16], 64);
        c0 = wmma_bf16(a, b0, c0);
        c1 = wmma_bf16(a, b1, c1);
        __syncthreads();
    }
#endif
    const int colL  = lane & 15;
    const int rhalf = (lane >> 4) * 8;
#pragma unroll
    for (int r = 0; r < 8; ++r) {
        int m = m0 + wm + r + rhalf;
        int n = n0 + wn + colL;
        float v0 = c0[r] + bias[m];
        float v1 = c1[r] + bias[m];
        size_t base = (size_t)bidx * M * N + (size_t)m * N + n;
        if (OUTF32) {
            Cf[base]      = v0;
            Cf[base + 16] = v1;
        } else {
            Cb[base]      = f2bf(v0);
            Cb[base + 16] = f2bf(v1);
        }
    }
}

// ---------------------------------------------------------------------------
// LePE: depthwise 5x5 conv (pad 2) over V = qkv[:, 2E:].  One thread/pixel.
// ---------------------------------------------------------------------------
__global__ __launch_bounds__(256) void lepe_kernel(const unsigned short* __restrict__ qkv,
                                                   const float* __restrict__ w,
                                                   const float* __restrict__ bias,
                                                   unsigned short* __restrict__ out) {
    int gid = blockIdx.x * 256 + threadIdx.x; // (b*384 + c)*4096 + p
    int p = gid & 4095;
    int c = (gid >> 12) % EDIM;
    int b = gid / (EDIM * LTOK);
    int y = p >> 6, x = p & 63;
    const unsigned short* v = qkv + ((size_t)b * E3 + 2 * EDIM + c) * LTOK;
    const float* wc = w + c * 25;
    float acc = bias[c];
#pragma unroll
    for (int ky = 0; ky < 5; ++ky) {
        int iy = y + ky - 2;
        if (iy < 0 || iy >= 64) continue;
#pragma unroll
        for (int kx = 0; kx < 5; ++kx) {
            int ix = x + kx - 2;
            if (ix < 0 || ix >= 64) continue;
            acc += bf2f(v[iy * 64 + ix]) * wc[ky * 5 + kx];
        }
    }
    out[gid] = f2bf(acc);
}

// ---------------------------------------------------------------------------
// Cluster stats: cos(token) = <cls_dir, k_tok> / ||k_tok||.  One block/(b,nh).
// (SCALE cancels in the cosine, so it is omitted here.)
// ---------------------------------------------------------------------------
__global__ __launch_bounds__(256) void stats_kernel(const unsigned short* __restrict__ qkv,
                                                    float* __restrict__ cosout) {
    int bn = blockIdx.x;
    int b = bn / NHEAD, nh = bn % NHEAD;
    const unsigned short* kb = qkv + ((size_t)b * E3 + EDIM + nh * HDIM) * LTOK;
    __shared__ float cls[HDIM];
    __shared__ float clsn[HDIM];
    int tid = threadIdx.x;
    if (tid < HDIM) cls[tid] = 0.f;
    __syncthreads();
    int hd = tid & 31, grp = tid >> 5; // 8 partial sums per channel
    float s = 0.f;
    for (int t = grp; t < LTOK; t += 8) s += bf2f(kb[(size_t)hd * LTOK + t]);
    atomicAdd(&cls[hd], s); // ds_add_f32
    __syncthreads();
    if (tid == 0) {
        float n2 = 0.f;
        for (int i = 0; i < HDIM; ++i) n2 += cls[i] * cls[i];
        float inv = 1.f / fmaxf(sqrtf(n2), 1e-12f);
        for (int i = 0; i < HDIM; ++i) clsn[i] = cls[i] * inv;
    }
    __syncthreads();
    for (int t = tid; t < LTOK; t += 256) {
        float dot = 0.f, n2 = 0.f;
#pragma unroll
        for (int i = 0; i < HDIM; ++i) {
            float kv = bf2f(kb[(size_t)i * LTOK + t]);
            dot += clsn[i] * kv;
            n2  += kv * kv;
        }
        cosout[(size_t)bn * LTOK + t] = dot / fmaxf(sqrtf(n2), 1e-12f);
    }
}

// ---------------------------------------------------------------------------
// Bitonic argsort (ascending) of 4096 cos values. One 1024-thread block/(b,nh).
// ---------------------------------------------------------------------------
__global__ __launch_bounds__(1024) void sort_kernel(const float* __restrict__ cosin,
                                                    int* __restrict__ idxout) {
    __shared__ float vals[LTOK];
    __shared__ int   inds[LTOK];
    int bn = blockIdx.x, tid = threadIdx.x;
    for (int i = tid; i < LTOK; i += 1024) {
        vals[i] = cosin[(size_t)bn * LTOK + i];
        inds[i] = i;
    }
    __syncthreads();
    for (int k = 2; k <= LTOK; k <<= 1) {
        for (int j = k >> 1; j > 0; j >>= 1) {
#pragma unroll
            for (int s = 0; s < 4; ++s) {
                int i = tid + s * 1024;
                int ixj = i ^ j;
                if (ixj > i) {
                    bool up = ((i & k) == 0);
                    float vi = vals[i], vj = vals[ixj];
                    if ((vi > vj) == up) {
                        vals[i] = vj; vals[ixj] = vi;
                        int t = inds[i]; inds[i] = inds[ixj]; inds[ixj] = t;
                    }
                }
            }
            __syncthreads();
        }
    }
    for (int i = tid; i < LTOK; i += 1024) idxout[(size_t)bn * LTOK + i] = inds[i];
}

// ---------------------------------------------------------------------------
// Chunked attention, flash-style with WMMA. One block/(chunk, head, batch),
// 512 threads = 16 waves; wave w owns query rows [w*16, w*16+16).
// ks is kept [key][hd] and vs transposed [hd][key] so B-fragments are
// K-contiguous per lane (wide dword LDS reads).
// Fuses the inverse permutation scatter + LePE add + bf16 convert.
// ---------------------------------------------------------------------------
__global__ __launch_bounds__(512) void attn_kernel(const unsigned short* __restrict__ qkv,
                                                   const int* __restrict__ idxg,
                                                   const unsigned short* __restrict__ lepe,
                                                   unsigned short* __restrict__ res) {
    __shared__ unsigned short qs [LCHNK * HDIM];     // [row][hd]
    __shared__ unsigned short ks [LCHNK * HDIM];     // [key][hd]  (k * SCALE)
    __shared__ unsigned short vsT[HDIM * LCHNK];     // [hd][key]
    __shared__ unsigned short pst[16 * 16 * HDIM];   // per-wave P staging 16x32
    int chunk = blockIdx.x, nh = blockIdx.y, b = blockIdx.z;
    int tid = threadIdx.x;
    const int* idxp = idxg + (size_t)(b * NHEAD + nh) * LTOK + chunk * LCHNK;
    const unsigned short* qb = qkv + ((size_t)b * E3 + nh * HDIM) * LTOK;
    const unsigned short* kb = qb + (size_t)EDIM * LTOK;
    const unsigned short* vb = qb + (size_t)(2 * EDIM) * LTOK;

    // gather sorted tokens into LDS
    for (int e = tid; e < LCHNK * HDIM; e += 512) {
        int row = e >> 5, hd = e & 31;
        int tok = idxp[row];
        qs[row * HDIM + hd]   = qb[(size_t)hd * LTOK + tok];
        ks[row * HDIM + hd]   = f2bf(bf2f(kb[(size_t)hd * LTOK + tok]) * SCALE_K);
        vsT[hd * LCHNK + row] = vb[(size_t)hd * LTOK + tok];
    }
    __syncthreads();

    int wave = tid >> 5, lane = tid & 31;
    int colL = lane & 15, half = lane >> 4;
    v16bf qfrag = load_a_frag(&qs[wave * 16 * HDIM], HDIM);
    unsigned short* myp = &pst[wave * 16 * HDIM];

    float rmax[8], rsum[8];
    v8f oc0 = {}, oc1 = {};
#pragma unroll
    for (int r = 0; r < 8; ++r) { rmax[r] = -1e30f; rsum[r] = 0.f; }
    v8f zf = {};

    for (int g = 0; g < 8; ++g) { // 8 groups of 32 keys
        // S = q @ k^T : B[k=hd][n=key] read from ks[n=key][k=hd]
        v16bf kf0 = load_b_frag_nk(&ks[(g * 32) * HDIM], HDIM);
        v16bf kf1 = load_b_frag_nk(&ks[(g * 32 + 16) * HDIM], HDIM);
        v8f s0 = wmma_bf16(qfrag, kf0, zf); // cols = keys g*32+0..15
        v8f s1 = wmma_bf16(qfrag, kf1, zf); // cols = keys g*32+16..31
#pragma unroll
        for (int r = 0; r < 8; ++r) {
            float a = s0[r], c = s1[r];
            float m = fmaxf(a, c);
            m = fmaxf(m, __shfl_xor(m, 1, 32));
            m = fmaxf(m, __shfl_xor(m, 2, 32));
            m = fmaxf(m, __shfl_xor(m, 4, 32));
            m = fmaxf(m, __shfl_xor(m, 8, 32));
            float nm    = fmaxf(rmax[r], m);
            float alpha = __expf(rmax[r] - nm);
            float p0 = __expf(a - nm);
            float p1 = __expf(c - nm);
            float ps = p0 + p1;
            ps += __shfl_xor(ps, 1, 32);
            ps += __shfl_xor(ps, 2, 32);
            ps += __shfl_xor(ps, 4, 32);
            ps += __shfl_xor(ps, 8, 32);
            rsum[r] = rsum[r] * alpha + ps;
            rmax[r] = nm;
            oc0[r] *= alpha;
            oc1[r] *= alpha;
            int mr = r + 8 * half;
            myp[mr * HDIM + colL]      = f2bf(p0);
            myp[mr * HDIM + 16 + colL] = f2bf(p1);
        }
        v16bf pf = load_a_frag(myp, HDIM);
        // O += P @ V : B[k=key][n=hd] read from vsT[n=hd][k=key]
        v16bf vf0 = load_b_frag_nk(&vsT[g * 32], LCHNK);
        v16bf vf1 = load_b_frag_nk(&vsT[16 * LCHNK + g * 32], LCHNK);
        oc0 = wmma_bf16(pf, vf0, oc0); // hd 0..15
        oc1 = wmma_bf16(pf, vf1, oc1); // hd 16..31
    }

    const unsigned short* lepeb = lepe + (size_t)b * EDIM * LTOK;
    unsigned short* resb = res + (size_t)b * EDIM * LTOK;
#pragma unroll
    for (int r = 0; r < 8; ++r) {
        float inv = 1.f / rsum[r];
        int mr  = r + 8 * half;
        int tok = idxp[wave * 16 + mr]; // scatter back through the permutation
        int ch0 = nh * HDIM + colL;
        int ch1 = ch0 + 16;
        float v0 = oc0[r] * inv + bf2f(lepeb[(size_t)ch0 * LTOK + tok]);
        float v1 = oc1[r] * inv + bf2f(lepeb[(size_t)ch1 * LTOK + tok]);
        resb[(size_t)ch0 * LTOK + tok] = f2bf(v0);
        resb[(size_t)ch1 * LTOK + tok] = f2bf(v1);
    }
}

// ---------------------------------------------------------------------------
// Host launch
// ---------------------------------------------------------------------------
extern "C" void kernel_launch(void* const* d_in, const int* in_sizes, int n_in,
                              void* d_out, int out_size, void* d_ws, size_t ws_size,
                              hipStream_t stream) {
    const float* x      = (const float*)d_in[0];
    const float* qkv_w  = (const float*)d_in[1];
    const float* qkv_b  = (const float*)d_in[2];
    const float* lepe_w = (const float*)d_in[3];
    const float* lepe_b = (const float*)d_in[4];
    const float* out_w  = (const float*)d_in[5];
    const float* out_b  = (const float*)d_in[6];
    float* out = (float*)d_out;

    char* ws = (char*)d_ws;
    size_t off = 0;
    auto take = [&](size_t bytes) -> void* {
        void* p = ws + off;
        off += (bytes + 255) & ~(size_t)255;
        return p;
    };
    unsigned short* x_bf    = (unsigned short*)take((size_t)BATCH * EDIM * LTOK * 2);
    unsigned short* qkvw_bf = (unsigned short*)take((size_t)E3 * EDIM * 2);
    unsigned short* outw_bf = (unsigned short*)take((size_t)EDIM * EDIM * 2);
    unsigned short* qkv_bf  = (unsigned short*)take((size_t)BATCH * E3 * LTOK * 2);
    unsigned short* lepe_bf = (unsigned short*)take((size_t)BATCH * EDIM * LTOK * 2);
    float*          cosb    = (float*)take((size_t)BATCH * NHEAD * LTOK * 4);
    int*            idxb    = (int*)take((size_t)BATCH * NHEAD * LTOK * 4);
    unsigned short* res_bf  = (unsigned short*)take((size_t)BATCH * EDIM * LTOK * 2);

    // 1) bf16 conversions
    {
        int n = BATCH * EDIM * LTOK;
        cvt_bf16_kernel<<<(n + 255) / 256, 256, 0, stream>>>(x, x_bf, n);
        n = E3 * EDIM;
        cvt_bf16_kernel<<<(n + 255) / 256, 256, 0, stream>>>(qkv_w, qkvw_bf, n);
        n = EDIM * EDIM;
        cvt_bf16_kernel<<<(n + 255) / 256, 256, 0, stream>>>(out_w, outw_bf, n);
    }
    // 2) QKV projection (bf16 WMMA GEMM, TDM double-buffered tiles)
    gemm_bf16_kernel<false><<<dim3(E3 / 64, LTOK / 64, BATCH), 256, 0, stream>>>(
        qkvw_bf, x_bf, qkv_b, nullptr, qkv_bf, E3, LTOK, EDIM);
    // 3) LePE depthwise conv on V
    {
        int n = BATCH * EDIM * LTOK;
        lepe_kernel<<<n / 256, 256, 0, stream>>>(qkv_bf, lepe_w, lepe_b, lepe_bf);
    }
    // 4) cosine-vs-cluster-center stats
    stats_kernel<<<BATCH * NHEAD, 256, 0, stream>>>(qkv_bf, cosb);
    // 5) argsort per (b, head)
    sort_kernel<<<BATCH * NHEAD, 1024, 0, stream>>>(cosb, idxb);
    // 6) chunked attention (WMMA flash) + un-permute + LePE add
    attn_kernel<<<dim3(NCHNK, NHEAD, BATCH), 512, 0, stream>>>(qkv_bf, idxb, lepe_bf, res_bf);
    // 7) output projection (bf16 WMMA GEMM, f32 out)
    gemm_bf16_kernel<true><<<dim3(EDIM / 64, LTOK / 64, BATCH), 256, 0, stream>>>(
        outw_bf, res_bf, out_b, out, nullptr, EDIM, LTOK, EDIM);

    (void)in_sizes; (void)n_in; (void)out_size; (void)ws_size;
}
